// TransformerModel_64982855189104
// MI455X (gfx1250) — matrix operand order repaired and checked by reading, not compiled, not run on hardware
//
#include <hip/hip_runtime.h>
#include <hip/hip_bf16.h>
#include <math.h>
#include <stdint.h>

// ---------------- problem constants ----------------
#define Bc   32
#define Vc   511
#define Sc   512            // V + 1 (cls)
#define Dc   512
#define Hc   8
#define DHc  64             // D / H
#define FFc  2048
#define Lc   4
#define Cc   30
#define TDc  1536           // 3*D
#define BSc  (Bc * Sc)      // 16384 rows of the residual stream
#define BVc  (Bc * Vc)      // 16352 tokens before cls prepend

typedef _Float16 half_t;
typedef __attribute__((ext_vector_type(16))) _Float16 v16h;
typedef __attribute__((ext_vector_type(8)))  _Float16 h8;
typedef __attribute__((ext_vector_type(4)))  _Float16 h4;
typedef __attribute__((ext_vector_type(8)))  float    v8f;
typedef __attribute__((ext_vector_type(4)))  float    f4;
typedef __attribute__((ext_vector_type(4)))  uint32_t u32x4;
typedef __attribute__((ext_vector_type(8)))  uint32_t u32x8;

// Async direct-to-LDS copy (CDNA5 GLOBAL_LOAD_ASYNC_TO_LDS_*, ASYNCcnt).
// LDS operand: low 32 bits of the generic pointer == workgroup LDS offset.
__device__ __forceinline__ void async_ld_lds_b64(const half_t* gsrc, half_t* ldst) {
  uint32_t l = (uint32_t)(uintptr_t)ldst;
  uint64_t g = (uint64_t)(uintptr_t)gsrc;
  asm volatile("global_load_async_to_lds_b64 %0, %1, off" :: "v"(l), "v"(g) : "memory");
}
__device__ __forceinline__ void wait_async_zero() {
  asm volatile("s_wait_asynccnt 0x0" ::: "memory");
}

// ---------------------------------------------------------------------------
// f32 -> f16 conversion (weights)
// ---------------------------------------------------------------------------
__global__ __launch_bounds__(256) void convert_f32_f16_kernel(
    const float* __restrict__ in, half_t* __restrict__ out, int n) {
  int i = blockIdx.x * 256 + threadIdx.x;
  if (i < n) out[i] = (half_t)in[i];
}

// ---------------------------------------------------------------------------
// Input MLP layer 1: (B*V,3) @ (3,1024) + b, ReLU -> f16.  K=3: scalar kernel.
// ---------------------------------------------------------------------------
__global__ __launch_bounds__(256) void mlp1_kernel(
    const float* __restrict__ src, const float* __restrict__ w1,
    const float* __restrict__ b1, half_t* __restrict__ out) {
  size_t idx = (size_t)blockIdx.x * 256 + threadIdx.x;   // < BVc*1024
  int row = (int)(idx >> 10);
  int col = (int)(idx & 1023);
  float a = b1[col];
  a += src[row * 3 + 0] * w1[0 * 1024 + col];
  a += src[row * 3 + 1] * w1[1 * 1024 + col];
  a += src[row * 3 + 2] * w1[2 * 1024 + col];
  out[idx] = (half_t)fmaxf(a, 0.f);
}

// ---------------------------------------------------------------------------
// Generic WMMA GEMM:  C(MxN) = A(MxK,f16) * B(KxN,f16) + bias, optional ReLU.
// Block = 256 threads (8 waves). Block tile 32(M) x 256(N), K-step 32.
// A tile: async direct-to-LDS (ASYNCcnt).  B tile: Tensor Data Mover
// (tensor_load_to_lds, TENSORcnt), issued once per K-step by wave 0.
// Epilogue restaged through LDS for coalesced b128 global stores.
// ---------------------------------------------------------------------------
template <bool RELU, bool OUT_F16>
__global__ __launch_bounds__(256) void gemm_bias_kernel(
    const half_t* __restrict__ A, const half_t* __restrict__ Bm,
    const float* __restrict__ bias, void* __restrict__ Cout,
    int M, int N, int K) {
  __shared__ half_t As[32 * 32];
  __shared__ float  Stage[32 * 256];   // 32 KB; first 16 KB doubles as B tile
  half_t* Bs = reinterpret_cast<half_t*>(Stage);

  const int t     = threadIdx.x;
  const int wave  = t >> 5;
  const int lane  = t & 31;
  const int l16   = lane & 15;
  const int lhalf = lane >> 4;     // 0 or 1
  const int m0 = blockIdx.y * 32;
  const int n0 = blockIdx.x * 256;

  v8f acc[2][2];
  for (int i = 0; i < 2; ++i)
    for (int j = 0; j < 2; ++j) acc[i][j] = {};

  const int ar = t >> 3, ac = (t & 7) * 4;   // A-tile stage coords (b64 each)

  // --- TDM descriptor group 1 (uniform across the K loop) ---
  // data_size=2B; tensor_dim0=N, tensor_dim1=K; tile 256x32; dim0_stride=N.
  u32x8 g1;
  g1[0] = 0x10000u;                                   // data_size = 1 -> 2 bytes
  g1[1] = ((uint32_t)N & 0xFFFFu) << 16;              // tensor_dim0 lo16
  g1[2] = ((uint32_t)N >> 16) | (((uint32_t)K & 0xFFFFu) << 16);
  g1[3] = (((uint32_t)K >> 16) & 0xFFFFu) | (256u << 16);   // tile_dim0 = 256
  g1[4] = 32u;                                        // tile_dim1 = 32, tile_dim2 = 0
  g1[5] = (uint32_t)N;                                // tensor_dim0_stride lo32
  g1[6] = 0u;                                         // stride hi, dim1_stride lo
  g1[7] = 0u;
  const uint32_t ldsB = (uint32_t)(uintptr_t)Bs;

  for (int k0 = 0; k0 < K; k0 += 32) {
    // stage A tile 32x32: one async b64 per thread
    async_ld_lds_b64(A + (size_t)(m0 + ar) * K + k0 + ac, &As[ar * 32 + ac]);
    // stage B tile 32x256 via Tensor Data Mover (wave 0 only)
    if (wave == 0) {
      uint64_t ga = (uint64_t)(uintptr_t)(Bm + (size_t)k0 * N + n0);
      u32x4 g0;
      g0[0] = 1u;                                     // count = 1 (valid D#)
      g0[1] = ldsB;                                   // lds_addr
      g0[2] = (uint32_t)ga;                           // global_addr lo32
      g0[3] = ((uint32_t)(ga >> 32) & 0x1FFFFFFu) | (2u << 30);  // addr hi, type=2
      asm volatile("tensor_load_to_lds %0, %1" :: "s"(g0), "s"(g1) : "memory");
      __builtin_amdgcn_s_wait_tensorcnt(0);
    }
    wait_async_zero();
    __syncthreads();

    v16h af[2], bf[2];
#pragma unroll
    for (int i = 0; i < 2; ++i) {
      int m = i * 16 + l16;
      int kb = lhalf * 16;
#pragma unroll
      for (int e = 0; e < 8; ++e) {
        af[i][e]     = As[m * 32 + kb + e];
        af[i][8 + e] = As[m * 32 + kb + 8 + e];
      }
    }
#pragma unroll
    for (int j = 0; j < 2; ++j) {
      int n = wave * 32 + j * 16 + l16;
      int kb2 = lhalf * 16;
#pragma unroll
      for (int e = 0; e < 16; ++e) bf[j][e] = Bs[(kb2 + e) * 256 + n];
    }
#pragma unroll
    for (int i = 0; i < 2; ++i)
#pragma unroll
      for (int j = 0; j < 2; ++j)
        acc[i][j] = __builtin_amdgcn_wmma_f32_16x16x32_f16(
            false, af[i], false, bf[j], (short)0, acc[i][j], false, false);
    __syncthreads();
  }

  // ---- epilogue: bias (+ReLU), restage in LDS, coalesced b128 stores ----
  if (OUT_F16) {
#pragma unroll
    for (int i = 0; i < 2; ++i)
#pragma unroll
      for (int j = 0; j < 2; ++j) {
        int col_l = wave * 32 + j * 16 + l16;
        float bv = bias ? bias[n0 + col_l] : 0.f;
#pragma unroll
        for (int r = 0; r < 8; ++r) {
          int row_l = i * 16 + lhalf * 8 + r;
          float v = acc[i][j][r] + bv;
          if (RELU) v = fmaxf(v, 0.f);
          Bs[row_l * 256 + col_l] = (half_t)v;
        }
      }
    __syncthreads();
#pragma unroll
    for (int it = 0; it < 4; ++it) {
      int lin = (it * 256 + t) * 8;
      int r = lin >> 8, c = lin & 255;
      *reinterpret_cast<h8*>((half_t*)Cout + (size_t)(m0 + r) * N + n0 + c) =
          *reinterpret_cast<h8*>(&Bs[lin]);
    }
  } else {
#pragma unroll
    for (int i = 0; i < 2; ++i)
#pragma unroll
      for (int j = 0; j < 2; ++j) {
        int col_l = wave * 32 + j * 16 + l16;
        float bv = bias ? bias[n0 + col_l] : 0.f;
#pragma unroll
        for (int r = 0; r < 8; ++r) {
          int row_l = i * 16 + lhalf * 8 + r;
          float v = acc[i][j][r] + bv;
          if (RELU) v = fmaxf(v, 0.f);
          Stage[row_l * 256 + col_l] = v;
        }
      }
    __syncthreads();
#pragma unroll
    for (int it = 0; it < 8; ++it) {
      int lin = (it * 256 + t) * 4;
      int r = lin >> 8, c = lin & 255;
      *reinterpret_cast<f4*>((float*)Cout + (size_t)(m0 + r) * N + n0 + c) =
          *reinterpret_cast<f4*>(&Stage[lin]);
    }
  }
}

// ---------------------------------------------------------------------------
// Attention scores: per (b,h): S x S = Q(Sx64) * K^T, scale + pad mask -> f16.
// B operand (K^T): per key index the 64 dot-dim values are contiguous ->
// two h8 loads per fragment.  Output restaged in LDS for b128 stores.
// Block 256 thr (8 waves) -> 32(q) x 256(k) tile; grid (S/256, S/32, B*H).
// ---------------------------------------------------------------------------
__global__ __launch_bounds__(256) void attn_scores_kernel(
    const half_t* __restrict__ qkv, const unsigned char* __restrict__ pad,
    half_t* __restrict__ scores) {
  __shared__ half_t st[32 * 256];
  const int t = threadIdx.x, wave = t >> 5, lane = t & 31;
  const int l16 = lane & 15, lhalf = lane >> 4;
  const int q0 = blockIdx.y * 32;
  const int n0 = blockIdx.x * 256;
  const int bh = blockIdx.z;
  const int b = bh >> 3, h = bh & 7;

  const half_t* qbase = qkv + (size_t)b * Sc * TDc + (size_t)h * DHc;
  const half_t* kbase = qbase + Dc;   // K section at column offset D

  v8f acc[2][2];
  for (int i = 0; i < 2; ++i)
    for (int j = 0; j < 2; ++j) acc[i][j] = {};

  for (int k0 = 0; k0 < DHc; k0 += 32) {
    v16h af[2], bf[2];
#pragma unroll
    for (int i = 0; i < 2; ++i) {
      int m = q0 + i * 16 + l16;
      int kb = lhalf * 16;
      h8 lo = *reinterpret_cast<const h8*>(qbase + (size_t)m * TDc + k0 + kb);
      h8 hi = *reinterpret_cast<const h8*>(qbase + (size_t)m * TDc + k0 + kb + 8);
#pragma unroll
      for (int e = 0; e < 8; ++e) { af[i][e] = lo[e]; af[i][8 + e] = hi[e]; }
    }
#pragma unroll
    for (int j = 0; j < 2; ++j) {
      int n = n0 + wave * 32 + j * 16 + l16;   // key index
      int kb2 = lhalf * 16;
      const h8* p = reinterpret_cast<const h8*>(kbase + (size_t)n * TDc + k0 + kb2);
      h8 c0 = p[0], c1 = p[1];
#pragma unroll
      for (int e = 0; e < 8; ++e) { bf[j][e] = c0[e]; bf[j][8 + e] = c1[e]; }
    }
#pragma unroll
    for (int i = 0; i < 2; ++i)
#pragma unroll
      for (int j = 0; j < 2; ++j)
        acc[i][j] = __builtin_amdgcn_wmma_f32_16x16x32_f16(
            false, af[i], false, bf[j], (short)0, acc[i][j], false, false);
  }

#pragma unroll
  for (int i = 0; i < 2; ++i)
#pragma unroll
    for (int j = 0; j < 2; ++j) {
      int col_l = wave * 32 + j * 16 + l16;
      int col = n0 + col_l;
      bool masked = (col > 0) && (pad[(size_t)b * Vc + col - 1] != 0);
#pragma unroll
      for (int r = 0; r < 8; ++r) {
        int row_l = i * 16 + lhalf * 8 + r;
        float v = acc[i][j][r] * 0.125f;      // 1/sqrt(dh), dh=64
        if (masked) v = -1.0e30f;             // -> f16 -inf
        st[row_l * 256 + col_l] = (half_t)v;
      }
    }
  __syncthreads();
#pragma unroll
  for (int it = 0; it < 4; ++it) {
    int lin = (it * 256 + t) * 8;
    int r = lin >> 8, c = lin & 255;
    *reinterpret_cast<h8*>(scores + ((size_t)bh * Sc + q0 + r) * Sc + n0 + c) =
        *reinterpret_cast<h8*>(&st[lin]);
  }
}

// ---------------------------------------------------------------------------
// Fused softmax over keys + head-mean into attn output (f16 probs in place).
// One block per (b,q); loops the 8 heads sequentially (no inter-block race).
// ---------------------------------------------------------------------------
__global__ __launch_bounds__(256) void softmax_attnmean_kernel(
    half_t* __restrict__ scores, float* __restrict__ attn_out, int layer) {
  const int bq = blockIdx.x;
  const int b = bq / Sc, q = bq % Sc;
  const int t = threadIdx.x;
  __shared__ float red[256];
  float acc0 = 0.f, acc1 = 0.f;

  for (int h = 0; h < Hc; ++h) {
    half_t* row = scores + (((size_t)(b * Hc + h)) * Sc + q) * Sc;
    float v0 = (float)row[t];
    float v1 = (float)row[t + 256];
    red[t] = fmaxf(v0, v1);
    __syncthreads();
    for (int s = 128; s > 0; s >>= 1) {
      if (t < s) red[t] = fmaxf(red[t], red[t + s]);
      __syncthreads();
    }
    float mx = red[0];
    __syncthreads();
    float e0 = __expf(v0 - mx), e1 = __expf(v1 - mx);
    red[t] = e0 + e1;
    __syncthreads();
    for (int s = 128; s > 0; s >>= 1) {
      if (t < s) red[t] += red[t + s];
      __syncthreads();
    }
    float inv = 1.f / red[0];
    __syncthreads();
    float p0 = e0 * inv, p1 = e1 * inv;
    row[t] = (half_t)p0;
    row[t + 256] = (half_t)p1;
    acc0 += p0 * 0.125f;   // 1/H
    acc1 += p1 * 0.125f;
  }
  float* out = attn_out + (((size_t)(b * Lc + layer)) * Sc + q) * Sc;
  out[t] = acc0;
  out[t + 256] = acc1;
}

// ---------------------------------------------------------------------------
// V transpose (LDS-tiled): vt[b][h][n][s] = qkv[b*S+s][2D + h*64 + n].
// Coalesced h8 reads and writes; padded LDS tile kills bank conflicts.
// Grid (S/32, B*H), block 256.
// ---------------------------------------------------------------------------
__global__ __launch_bounds__(256) void v_transpose_kernel(
    const half_t* __restrict__ qkv, half_t* __restrict__ vt) {
  __shared__ half_t tile[DHc][33];
  const int s0 = blockIdx.x * 32;
  const int bh = blockIdx.y;
  const int b = bh >> 3, h = bh & 7;
  const int t = threadIdx.x;
  {
    int sl = t >> 3;            // 0..31  (s within tile)
    int nq = (t & 7) * 8;       // 0..56  (n chunk)
    h8 v = *reinterpret_cast<const h8*>(
        qkv + ((size_t)(b * Sc + s0 + sl)) * TDc + 2 * Dc + h * DHc + nq);
#pragma unroll
    for (int e = 0; e < 8; ++e) tile[nq + e][sl] = v[e];
  }
  __syncthreads();
  {
    int nl = t >> 2;            // 0..63  (n within tile)
    int sq = (t & 3) * 8;       // 0..24  (s chunk)
    h8 v;
#pragma unroll
    for (int e = 0; e < 8; ++e) v[e] = tile[nl][sq + e];
    *reinterpret_cast<h8*>(vt + ((size_t)bh * DHc + nl) * Sc + s0 + sq) = v;
  }
}

// ---------------------------------------------------------------------------
// P @ V per (b,h): (SxS) * (Sx64) -> o[:, h*64:(h+1)*64] (f16).
// Both operands load as contiguous h8 pairs (V pre-transposed).
// Output restaged in LDS for b128 stores.
// Block = 64 thr (2 waves); grid (S/32, B*H).
// ---------------------------------------------------------------------------
__global__ __launch_bounds__(64) void attn_av_kernel(
    const half_t* __restrict__ probs, const half_t* __restrict__ vt,
    half_t* __restrict__ o) {
  __shared__ half_t st[32 * 64];
  const int t = threadIdx.x, wave = t >> 5, lane = t & 31;
  const int l16 = lane & 15, lhalf = lane >> 4;
  const int q0 = blockIdx.x * 32;
  const int bh = blockIdx.y;
  const int b = bh >> 3, h = bh & 7;

  const half_t* p     = probs + (size_t)bh * Sc * Sc;
  const half_t* vbase = vt + (size_t)bh * DHc * Sc;   // [n][s], ld = S

  v8f acc[2][2];
  for (int i = 0; i < 2; ++i)
    for (int j = 0; j < 2; ++j) acc[i][j] = {};

  for (int k0 = 0; k0 < Sc; k0 += 32) {
    v16h af[2], bf[2];
#pragma unroll
    for (int i = 0; i < 2; ++i) {
      int m = q0 + i * 16 + l16;
      int kb = lhalf * 16;
      h8 lo = *reinterpret_cast<const h8*>(p + (size_t)m * Sc + k0 + kb);
      h8 hi = *reinterpret_cast<const h8*>(p + (size_t)m * Sc + k0 + kb + 8);
#pragma unroll
      for (int e = 0; e < 8; ++e) { af[i][e] = lo[e]; af[i][8 + e] = hi[e]; }
    }
#pragma unroll
    for (int j = 0; j < 2; ++j) {
      int n = wave * 32 + j * 16 + l16;     // head-local V column (0..63)
      int kb2 = lhalf * 16;
      const h8* vp = reinterpret_cast<const h8*>(vbase + (size_t)n * Sc + k0 + kb2);
      h8 c0 = vp[0], c1 = vp[1];
#pragma unroll
      for (int e = 0; e < 8; ++e) { bf[j][e] = c0[e]; bf[j][8 + e] = c1[e]; }
    }
#pragma unroll
    for (int i = 0; i < 2; ++i)
#pragma unroll
      for (int j = 0; j < 2; ++j)
        acc[i][j] = __builtin_amdgcn_wmma_f32_16x16x32_f16(
            false, af[i], false, bf[j], (short)0, acc[i][j], false, false);
  }

#pragma unroll
  for (int i = 0; i < 2; ++i)
#pragma unroll
    for (int j = 0; j < 2; ++j) {
      int col_l = wave * 32 + j * 16 + l16;
#pragma unroll
      for (int r = 0; r < 8; ++r) {
        int row_l = i * 16 + lhalf * 8 + r;
        st[row_l * 64 + col_l] = (half_t)acc[i][j][r];
      }
    }
  __syncthreads();
#pragma unroll
  for (int it = 0; it < 4; ++it) {
    int lin = (it * 64 + t) * 8;
    int r = lin >> 6, c = lin & 63;
    *reinterpret_cast<h8*>(o + (size_t)(b * Sc + q0 + r) * Dc + h * DHc + c) =
        *reinterpret_cast<h8*>(&st[lin]);
  }
}

// ---------------------------------------------------------------------------
// x = LayerNorm(x + res) * g + b ; writes fp32 stream and f16 mirror.
// ---------------------------------------------------------------------------
__global__ __launch_bounds__(256) void add_ln_kernel(
    float* __restrict__ x, half_t* __restrict__ xh,
    const float* __restrict__ res, const float* __restrict__ g,
    const float* __restrict__ bta) {
  const size_t row = blockIdx.x;
  const int t = threadIdx.x;
  __shared__ float red[256];
  float y0 = x[row * Dc + t] + res[row * Dc + t];
  float y1 = x[row * Dc + t + 256] + res[row * Dc + t + 256];
  red[t] = y0 + y1;
  __syncthreads();
  for (int s = 128; s > 0; s >>= 1) {
    if (t < s) red[t] += red[t + s];
    __syncthreads();
  }
  float mu = red[0] * (1.f / Dc);
  __syncthreads();
  float d0 = y0 - mu, d1 = y1 - mu;
  red[t] = d0 * d0 + d1 * d1;
  __syncthreads();
  for (int s = 128; s > 0; s >>= 1) {
    if (t < s) red[t] += red[t + s];
    __syncthreads();
  }
  float inv = rsqrtf(red[0] * (1.f / Dc) + 1e-5f);
  float o0 = d0 * inv * g[t] + bta[t];
  float o1 = d1 * inv * g[t + 256] + bta[t + 256];
  x[row * Dc + t] = o0;
  x[row * Dc + t + 256] = o1;
  xh[row * Dc + t] = (half_t)o0;
  xh[row * Dc + t + 256] = (half_t)o1;
}

// ---------------------------------------------------------------------------
// Assemble residual stream: [cls ; mlp_out] + sinusoidal PE (on the fly).
// ---------------------------------------------------------------------------
__global__ __launch_bounds__(256) void assemble_pe_kernel(
    const float* __restrict__ mlp_out, const float* __restrict__ cls,
    float* __restrict__ x, half_t* __restrict__ xh) {
  const int bs = blockIdx.x;
  const int b = bs / Sc, s = bs % Sc;
  const int t = threadIdx.x;
#pragma unroll
  for (int u = 0; u < 2; ++u) {
    int idx = t + u * 256;
    float v = (s == 0) ? cls[idx]
                       : mlp_out[((size_t)b * Vc + s - 1) * Dc + idx];
    int i2 = idx & ~1;
    float div = __expf(-(float)i2 * (9.210340372f / (float)Dc));  // ln(1e4)/D
    float ang = (float)s * div;
    v += (idx & 1) ? __cosf(ang) : __sinf(ang);
    x[(size_t)bs * Dc + idx] = v;
    xh[(size_t)bs * Dc + idx] = (half_t)v;
  }
}

// ---------------------------------------------------------------------------
// Classifier: cls rows @ fc_w (512x30) + fc_b -> d_out[0:B*C].
// ---------------------------------------------------------------------------
__global__ __launch_bounds__(32) void classifier_kernel(
    const float* __restrict__ x, const float* __restrict__ fcw,
    const float* __restrict__ fcb, float* __restrict__ out) {
  const int b = blockIdx.x;
  const int c = threadIdx.x;
  if (c >= Cc) return;
  const float* xr = x + (size_t)b * Sc * Dc;   // cls token = row 0 of batch b
  float a = fcb[c];
  for (int d = 0; d < Dc; ++d) a += xr[d] * fcw[d * Cc + c];
  out[b * Cc + c] = a;
}

// ---------------------------------------------------------------------------
// Host-side launch orchestration
// ---------------------------------------------------------------------------
extern "C" void kernel_launch(void* const* d_in, const int* in_sizes, int n_in,
                              void* d_out, int out_size, void* d_ws, size_t ws_size,
                              hipStream_t stream) {
  (void)in_sizes; (void)n_in; (void)out_size; (void)ws_size;

  const float* src      = (const float*)d_in[0];
  const unsigned char* pad = (const unsigned char*)d_in[1];
  const float* mlp_w1   = (const float*)d_in[2];
  const float* mlp_b1   = (const float*)d_in[3];
  const float* mlp_w2   = (const float*)d_in[4];
  const float* mlp_b2   = (const float*)d_in[5];
  const float* cls_tok  = (const float*)d_in[6];
  const float* qkv_w    = (const float*)d_in[7];
  const float* qkv_b    = (const float*)d_in[8];
  const float* out_w    = (const float*)d_in[9];
  const float* out_b    = (const float*)d_in[10];
  const float* ff1_w    = (const float*)d_in[11];
  const float* ff1_b    = (const float*)d_in[12];
  const float* ff2_w    = (const float*)d_in[13];
  const float* ff2_b    = (const float*)d_in[14];
  const float* ln1_g    = (const float*)d_in[15];
  const float* ln1_b    = (const float*)d_in[16];
  const float* ln2_g    = (const float*)d_in[17];
  const float* ln2_b    = (const float*)d_in[18];
  const float* fc_w     = (const float*)d_in[19];
  const float* fc_b     = (const float*)d_in[20];

  float* out_logits = (float*)d_out;                 // (B,C)
  float* attn_out   = (float*)d_out + Bc * Cc;       // (B,L,S,S)

  // ---- workspace carve (all 256B aligned) ----
  uint8_t* ws = (uint8_t*)d_ws;
  size_t off = 0;
  auto carve = [&](size_t bytes) -> void* {
    void* p = ws + off;
    off = (off + bytes + 255) & ~(size_t)255;
    return p;
  };
  half_t* qkvw16  = (half_t*)carve((size_t)Lc * Dc * TDc * 2);
  half_t* outw16  = (half_t*)carve((size_t)Lc * Dc * Dc * 2);
  half_t* ff1w16  = (half_t*)carve((size_t)Lc * Dc * FFc * 2);
  half_t* ff2w16  = (half_t*)carve((size_t)Lc * FFc * Dc * 2);
  half_t* mlpw2h  = (half_t*)carve((size_t)2 * Dc * Dc * 2);          // (1024,512)
  float*  xf32    = (float*) carve((size_t)BSc * Dc * 4);
  half_t* xf16    = (half_t*)carve((size_t)BSc * Dc * 2);
  half_t* qkv16   = (half_t*)carve((size_t)BSc * TDc * 2);
  half_t* s16     = (half_t*)carve((size_t)Bc * Hc * Sc * Sc * 2);    // scores->probs
  half_t* vt16    = (half_t*)carve((size_t)BSc * Dc * 2);             // V transposed
  half_t* o16     = (half_t*)carve((size_t)BSc * Dc * 2);
  float*  t32     = (float*) carve((size_t)BSc * Dc * 4);             // GEMM f32 out
  half_t* ff16    = (half_t*)carve((size_t)BSc * FFc * 2);            // also mlp1 out

  auto conv = [&](const float* fin, half_t* fout, int n) {
    convert_f32_f16_kernel<<<(n + 255) / 256, 256, 0, stream>>>(fin, fout, n);
  };
  conv(qkv_w, qkvw16, Lc * Dc * TDc);
  conv(out_w, outw16, Lc * Dc * Dc);
  conv(ff1_w, ff1w16, Lc * Dc * FFc);
  conv(ff2_w, ff2w16, Lc * FFc * Dc);
  conv(mlp_w2, mlpw2h, 2 * Dc * Dc);

  // ---- input MLP ----
  mlp1_kernel<<<(BVc * 1024) / 256, 256, 0, stream>>>(src, mlp_w1, mlp_b1, ff16);
  gemm_bias_kernel<false, false><<<dim3(Dc / 256, BVc / 32), 256, 0, stream>>>(
      ff16, mlpw2h, mlp_b2, t32, BVc, Dc, 2 * Dc);
  assemble_pe_kernel<<<BSc, 256, 0, stream>>>(t32, cls_tok, xf32, xf16);

  // ---- transformer layers ----
  for (int l = 0; l < Lc; ++l) {
    // QKV projection (f16 out, bias fused)
    gemm_bias_kernel<false, true><<<dim3(TDc / 256, BSc / 32), 256, 0, stream>>>(
        xf16, qkvw16 + (size_t)l * Dc * TDc, qkv_b + l * TDc, qkv16,
        BSc, TDc, Dc);
    // scores = Q K^T * scale, masked
    attn_scores_kernel<<<dim3(Sc / 256, Sc / 32, Bc * Hc), 256, 0, stream>>>(
        qkv16, pad, s16);
    // softmax (in place -> probs) + head-averaged attention output
    softmax_attnmean_kernel<<<BSc, 256, 0, stream>>>(s16, attn_out, l);
    // V transpose for dot-dim-contiguous AV loads
    v_transpose_kernel<<<dim3(Sc / 32, Bc * Hc), 256, 0, stream>>>(qkv16, vt16);
    // O = P V
    attn_av_kernel<<<dim3(Sc / 32, Bc * Hc), 64, 0, stream>>>(s16, vt16, o16);
    // out projection (f32 out), then residual + LN1
    gemm_bias_kernel<false, false><<<dim3(Dc / 256, BSc / 32), 256, 0, stream>>>(
        o16, outw16 + (size_t)l * Dc * Dc, out_b + l * Dc, t32, BSc, Dc, Dc);
    add_ln_kernel<<<BSc, 256, 0, stream>>>(xf32, xf16, t32,
                                           ln1_g + l * Dc, ln1_b + l * Dc);
    // FFN
    gemm_bias_kernel<true, true><<<dim3(FFc / 256, BSc / 32), 256, 0, stream>>>(
        xf16, ff1w16 + (size_t)l * Dc * FFc, ff1_b + l * FFc, ff16,
        BSc, FFc, Dc);
    gemm_bias_kernel<false, false><<<dim3(Dc / 256, BSc / 32), 256, 0, stream>>>(
        ff16, ff2w16 + (size_t)l * FFc * Dc, ff2_b + l * Dc, t32,
        BSc, Dc, FFc);
    add_ln_kernel<<<BSc, 256, 0, stream>>>(xf32, xf16, t32,
                                           ln2_g + l * Dc, ln2_b + l * Dc);
  }

  // ---- classifier on cls tokens ----
  classifier_kernel<<<Bc, 32, 0, stream>>>(xf32, fc_w, fc_b, out_logits);
}
